// CWTTransform_29626684408204
// MI455X (gfx1250) — matrix operand order repaired and checked by reading, not compiled, not run on hardware
//
#include <hip/hip_runtime.h>
#include <hip/hip_bf16.h>

// ---------------------------------------------------------------------------
// CWT (Morlet filter bank) on gfx1250 via v_wmma_f32_16x16x32_f16.
//   B=4, C=3 -> 12 channels; T=16384; 128 scales; K=763 (padded to 768);
//   reflect pad 381; hop 64 -> 256 frames. Output (4,3,128,256) f32.
//
// Compute-bound (~77 GFLOP vs ~3 MB unique bytes): GEMM on the matrix pipes.
//   D[s,t] = sum_k W[s,k] * x[t-381+k]  -> A = filters (M=128 x K=768, f16)
//                                          B = Toeplitz signal window (K x 64)
// One workgroup (8 waves) per (channel, 64-wide hop frame); hop-mean pooling
// is block-local. Wave w owns a 16-scale stripe: 4 N-tiles x 24 K-steps x
// {re,im} = 192 WMMAs/wave, f32 accumulation.
// ---------------------------------------------------------------------------

typedef __attribute__((ext_vector_type(16))) _Float16 v16h;
typedef __attribute__((ext_vector_type(8)))  float    v8f;

static constexpr int NCH     = 12;      // B*C
static constexpr int T_LEN   = 16384;
static constexpr int NSCALE  = 128;
static constexpr int KRAW    = 763;     // true filter length
static constexpr int KPAD    = 768;     // padded to 24*32
static constexpr int KSTEPS  = KPAD / 32;
static constexpr int PADL    = 381;     // K//2
static constexpr int HOP     = 64;
static constexpr int NFRAMES = T_LEN / HOP;   // 256
static constexpr int WIN     = HOP + KPAD;    // 832 samples per block window

union FragAB { v16h h; uint4 q[2]; unsigned int u[8]; };

static __device__ inline v8f wmma16x16x32(v16h a, v16h b, v8f c) {
    return __builtin_amdgcn_wmma_f32_16x16x32_f16(
        /*neg_a=*/false, a, /*neg_b=*/false, b,
        /*c_mod=*/(short)0, c, /*reuse_a=*/false, /*reuse_b=*/false);
}

static __device__ inline int reflect_idx(int g) {
    // jnp.pad mode='reflect'; pad(381) < T so one reflection suffices.
    if (g < 0)       g = -g;
    if (g >= T_LEN)  g = 2 * T_LEN - 2 - g;
    return g;
}

// ---------------- pre-pass: f32 -> f16 conversions into workspace ----------

__global__ void cwt_convert_filters(const float* __restrict__ wr,
                                    const float* __restrict__ wi,
                                    _Float16* __restrict__ fr,
                                    _Float16* __restrict__ fi) {
    int i = blockIdx.x * blockDim.x + threadIdx.x;
    if (i >= NSCALE * KPAD) return;
    int s = i / KPAD, k = i % KPAD;
    float vr = (k < KRAW) ? wr[s * KRAW + k] : 0.0f;
    float vi = (k < KRAW) ? wi[s * KRAW + k] : 0.0f;
    fr[i] = (_Float16)vr;
    fi[i] = (_Float16)vi;
}

__global__ void cwt_convert_signal(const float* __restrict__ x,
                                   _Float16* __restrict__ xh) {
    int i = blockIdx.x * blockDim.x + threadIdx.x;
    if (i >= NCH * T_LEN) return;
    xh[i] = (_Float16)x[i];
}

// ---------------- main kernel ----------------------------------------------
// grid: (NFRAMES, NCH), block: 256 threads (8 waves).

__global__ void __launch_bounds__(256)
cwt_wmma_kernel(const _Float16* __restrict__ xh,
                const _Float16* __restrict__ fr,
                const _Float16* __restrict__ fi,
                float* __restrict__ out) {
    __shared__ __align__(16) _Float16 win [WIN];   // x window
    __shared__ __align__(16) _Float16 win2[WIN];   // same, shifted by +1 half
    __shared__ float pool[NSCALE];

    const int frame = blockIdx.x;
    const int ch    = blockIdx.y;
    const int t0    = frame * HOP;
    const int tid   = threadIdx.x;

    // Stage reflect-padded window (and +1-shifted copy so odd lanes get
    // dword-aligned Toeplitz fragment reads).
    const _Float16* xc = xh + ch * T_LEN;
    for (int i = tid; i < WIN; i += 256) {
        win [i] = xc[reflect_idx(t0 - PADL + i)];
        win2[i] = xc[reflect_idx(t0 - PADL + i + 1)];
    }
    for (int i = tid; i < NSCALE; i += 256) pool[i] = 0.0f;
    __syncthreads();

    const int lane = tid & 31;
    const int wv   = tid >> 5;          // 0..7  -> M-tile (16 scales each)
    const int n    = lane & 15;         // B column / A row within tile
    const int hi   = lane >> 4;         // lane half selector

    // A fragments: row m, halves at K = kk*32 + hi*16 + grp*8 + (0..7)
    // -> two aligned uint4 chunks per fragment.
    const int m = wv * 16 + n;
    const uint4* pr = (const uint4*)fr;
    const uint4* pi = (const uint4*)fi;
    const int abase = m * (KPAD / 8) + hi * 2;      // uint4 units

    // B fragment base: B[k][n] = win[kk*32 + (nt+hi)*16 + n + h].
    // kk*32 and (nt+hi)*16 are even, so alignment parity == n&1, per-lane
    // constant -> hoist the win/win2 select out of the whole loop. Every
    // fragment address is srcLane + even offset (folds into DS offsets).
    const _Float16* srcLane = (n & 1) ? (win2 + (n - 1)) : (win + n);

    v8f cre[4], cim[4];
    #pragma unroll
    for (int nt = 0; nt < 4; ++nt) { cre[nt] = (v8f)0.0f; cim[nt] = (v8f)0.0f; }

    #pragma unroll 2
    for (int kk = 0; kk < KSTEPS; ++kk) {
        FragAB are, aim;
        are.q[0] = pr[abase + kk * 4];
        are.q[1] = pr[abase + kk * 4 + 1];
        aim.q[0] = pi[abase + kk * 4];
        aim.q[1] = pi[abase + kk * 4 + 1];

        // Load all four B fragments into distinct registers first, so the
        // 8 WMMAs below issue back-to-back with no intervening VALU writes
        // to their sources (one WAR hazard gap per K-step instead of four).
        FragAB b[4];
        #pragma unroll
        for (int nt = 0; nt < 4; ++nt) {
            const unsigned int* bp =
                (const unsigned int*)(srcLane + kk * 32 + (nt + hi) * 16);
            #pragma unroll
            for (int j = 0; j < 8; ++j) b[nt].u[j] = bp[j];
        }

        #pragma unroll
        for (int nt = 0; nt < 4; ++nt) cre[nt] = wmma16x16x32(are.h, b[nt].h, cre[nt]);
        #pragma unroll
        for (int nt = 0; nt < 4; ++nt) cim[nt] = wmma16x16x32(aim.h, b[nt].h, cim[nt]);
    }

    // Magnitude + hop-mean pooling. Lane holds (M = wv*16 + hi*8 + r, all nt/N).
    #pragma unroll
    for (int nt = 0; nt < 4; ++nt) {
        #pragma unroll
        for (int r = 0; r < 8; ++r) {
            float re = cre[nt][r], im = cim[nt][r];
            float mag = log1pf(sqrtf(re * re + im * im + 1e-8f));
            atomicAdd(&pool[wv * 16 + hi * 8 + r], mag);   // ds_add_f32
        }
    }
    __syncthreads();

    for (int s = tid; s < NSCALE; s += 256)
        out[(ch * NSCALE + s) * NFRAMES + frame] = pool[s] * (1.0f / HOP);
}

// ---------------- launch ----------------------------------------------------

extern "C" void kernel_launch(void* const* d_in, const int* in_sizes, int n_in,
                              void* d_out, int out_size, void* d_ws, size_t ws_size,
                              hipStream_t stream) {
    const float* x  = (const float*)d_in[0];            // (4,3,16384)
    const float* wr = (const float*)d_in[1];            // (128,1,763)
    const float* wi = (const float*)d_in[2];            // (128,1,763)
    // d_in[3] = hop_length (64), baked into the kernel constants.

    char* ws = (char*)d_ws;
    _Float16* fr = (_Float16*)(ws);
    _Float16* fi = (_Float16*)(ws + (size_t)NSCALE * KPAD * sizeof(_Float16));
    _Float16* xh = (_Float16*)(ws + (size_t)2 * NSCALE * KPAD * sizeof(_Float16));

    {   // filters: 128*768 elements
        int nelem = NSCALE * KPAD;
        cwt_convert_filters<<<(nelem + 255) / 256, 256, 0, stream>>>(wr, wi, fr, fi);
    }
    {   // signal: 12*16384 elements
        int nelem = NCH * T_LEN;
        cwt_convert_signal<<<(nelem + 255) / 256, 256, 0, stream>>>(x, xh);
    }

    dim3 grid(NFRAMES, NCH);
    cwt_wmma_kernel<<<grid, 256, 0, stream>>>(xh, fr, fi, (float*)d_out);
}
